// Model_84404697301739
// MI455X (gfx1250) — matrix operand order repaired and checked by reading, not compiled, not run on hardware
//
#include <hip/hip_runtime.h>

// ---------------------------------------------------------------------------
// Types for CDNA5 WMMA / TDM
// ---------------------------------------------------------------------------
typedef __attribute__((ext_vector_type(16))) __bf16       v16bf;
typedef __attribute__((ext_vector_type(8)))  float        v8f;
typedef __attribute__((ext_vector_type(8)))  unsigned int v8u;
typedef __attribute__((ext_vector_type(4)))  unsigned int v4u;
typedef __attribute__((ext_vector_type(8)))  int          v8i;
typedef __attribute__((ext_vector_type(4)))  int          v4i;

static __device__ __forceinline__ v8f wmma_bf16(v8u a, v8u b, v8f c) {
    return __builtin_amdgcn_wmma_f32_16x16x32_bf16(
        /*neg_a=*/false, __builtin_bit_cast(v16bf, a),
        /*neg_b=*/false, __builtin_bit_cast(v16bf, b),
        /*c_mod=*/(short)0, c, /*reuse_a=*/false, /*reuse_b=*/false);
}

static __device__ __forceinline__ unsigned short f32_to_bf16(float f) {
    unsigned int u = __float_as_uint(f);
    unsigned int r = 0x7FFFu + ((u >> 16) & 1u);
    return (unsigned short)((u + r) >> 16);
}
static __device__ __forceinline__ float bf16_to_f32(unsigned short h) {
    return __uint_as_float(((unsigned int)h) << 16);
}

// A-fragment (16x32 bf16), rows row-major with `stride` (elements).
// Layout: lanes 0-15 hold row m=lane, K = {c0..c0+15};
// lanes 16-31 hold K = {c0+16..c0+31}.
static __device__ __forceinline__ v8u
load_frag_A(const unsigned short* base, int stride, int r0, int c0) {
    int lane = threadIdx.x & 31;
    const unsigned short* p =
        base + (size_t)(r0 + (lane & 15)) * stride + c0 + ((lane >> 4) << 4);
    uint4 lo = *(const uint4*)p;
    uint4 hi = *(const uint4*)(p + 8);
    v8u r; r[0]=lo.x; r[1]=lo.y; r[2]=lo.z; r[3]=lo.w;
           r[4]=hi.x; r[5]=hi.y; r[6]=hi.z; r[7]=hi.w;
    return r;
}

// B-fragment (32x16 bf16) where B = Mrow^T, i.e. B[k][n] = Mrow[n][k] with
// Mrow row-major (stride elements). Per-lane data is 16 contiguous elements.
static __device__ __forceinline__ v8u
load_frag_BT(const unsigned short* base, int stride, int n0, int c0) {
    int lane = threadIdx.x & 31;
    const unsigned short* p =
        base + (size_t)(n0 + (lane & 15)) * stride + c0 + ((lane >> 4) << 4);
    uint4 lo = *(const uint4*)p;
    uint4 hi = *(const uint4*)(p + 8);
    v8u r; r[0]=lo.x; r[1]=lo.y; r[2]=lo.z; r[3]=lo.w;
           r[4]=hi.x; r[5]=hi.y; r[6]=hi.z; r[7]=hi.w;
    return r;
}

// B-fragment from K-pair-packed LDS: word[r][n] = B[2r][n] | B[2r+1][n]<<16.
static __device__ __forceinline__ v8u
load_frag_Bp(const unsigned int* base, int ldw, int r0, int n0) {
    int lane = threadIdx.x & 31;
    const unsigned int* p =
        base + (size_t)(r0 + ((lane >> 4) << 3)) * ldw + n0 + (lane & 15);
    v8u r;
#pragma unroll
    for (int v = 0; v < 8; ++v) r[v] = p[v * ldw];
    return r;
}

// B-fragment from row-major [K][N] bf16 storage; packs K-pairs on the fly.
static __device__ __forceinline__ v8u
load_frag_Brow(const unsigned short* base, int stride, int k0, int n0) {
    int lane = threadIdx.x & 31;
    const unsigned short* p =
        base + (size_t)(k0 + ((lane >> 4) << 4)) * stride + n0 + (lane & 15);
    v8u r;
#pragma unroll
    for (int v = 0; v < 8; ++v) {
        unsigned int lo = p[(2 * v) * stride];
        unsigned int hi = p[(2 * v + 1) * stride];
        r[v] = lo | (hi << 16);
    }
    return r;
}

// TDM: async 2-D tile copy global -> LDS (bf16 elements).
// D# built per CDNA5 ISA ch.8: group0 = {flags(count=1), lds_addr,
// global_addr[31:0], global_addr[56:32] | type=2<<30}; group1 packs
// data_size=2B, tensor_dim0/1, tile_dim0/1, tensor_dim0_stride.
static __device__ __forceinline__ void
tdm_load_tile_2d(const unsigned short* gsrc, unsigned rows, unsigned cols,
                 unsigned row_stride_elems, const unsigned short* lds_dst) {
    unsigned long long ga = (unsigned long long)(const void*)gsrc;
    unsigned lds_addr = (unsigned)(unsigned long long)(const void*)lds_dst;
    v4u g0;
    g0[0] = 1u;                                        // count=1 valid D#
    g0[1] = lds_addr;                                  // LDS byte address
    g0[2] = (unsigned)(ga & 0xFFFFFFFFu);              // global_addr lo
    g0[3] = (unsigned)((ga >> 32) & 0x01FFFFFFu) | (2u << 30);  // hi | type=2
    v8i g1;
    g1[0] = (int)(1u << 16);                           // data_size=1 (2 bytes)
    g1[1] = (int)((cols & 0xFFFFu) << 16);             // tensor_dim0 lo16
    g1[2] = (int)((rows & 0xFFFFu) << 16);             // dim0 hi16=0 | dim1 lo16
    g1[3] = (int)((cols & 0xFFFFu) << 16);             // dim1 hi16=0 | tile_dim0
    g1[4] = (int)(rows & 0xFFFFu);                     // tile_dim1 | tile_dim2=0
    g1[5] = (int)row_stride_elems;                     // tensor_dim0_stride lo32
    g1[6] = 0;                                         // stride hi16 | dim1_stride
    g1[7] = 0;
    v4i gz = {0, 0, 0, 0};
#if __clang_major__ >= 23
    v8i gz8 = {0, 0, 0, 0, 0, 0, 0, 0};
    __builtin_amdgcn_tensor_load_to_lds(g0, g1, gz, gz, gz8, 0);
#else
    __builtin_amdgcn_tensor_load_to_lds(g0, g1, gz, gz, 0);
#endif
}

// ---------------------------------------------------------------------------
// Elementwise / prep kernels
// ---------------------------------------------------------------------------
__global__ void cvt_bf16_kernel(const float* __restrict__ src,
                                unsigned short* __restrict__ dst, int n) {
    int i = blockIdx.x * 256 + threadIdx.x;
    if (i < n) dst[i] = f32_to_bf16(src[i]);
}

__global__ void prep_emb_kernel(const float* __restrict__ src,   // [129][64]
                                unsigned short* __restrict__ dst) { // [144][64]
    int i = blockIdx.x * 256 + threadIdx.x;
    if (i < 144 * 64) {
        int r = i >> 6, c = i & 63;
        dst[i] = (r < 129) ? f32_to_bf16(src[r * 64 + c]) : (unsigned short)0;
    }
}

__global__ void prep_qpos_kernel(const int* __restrict__ lens, int ns,
                                 int total, int* __restrict__ qpos) {
    __shared__ int starts[64];
    if (threadIdx.x == 0) {
        int acc = 0;
        for (int s = 0; s < ns && s < 64; ++s) { starts[s] = acc; acc += lens[s]; }
    }
    __syncthreads();
    for (int i = threadIdx.x; i < total; i += blockDim.x) {
        int q = i;
        for (int s = 0; s < ns && s < 64; ++s)
            if (i >= starts[s]) q = i - starts[s];
        qpos[i] = q;
    }
}

// ---------------------------------------------------------------------------
// bf16 WMMA GEMM: C[M,N] = A[M,K] @ W[K,N] (+bias) (+relu)
// Block: 256 threads = 8 waves, 128x64 tile, K-step 32.
// Each wave owns a 32x32 output patch -> 4 WMMAs per K-step (A/B frag reuse).
// ---------------------------------------------------------------------------
__global__ __launch_bounds__(256) void gemm_bf16_kernel(
    const unsigned short* __restrict__ A, int lda,
    const unsigned short* __restrict__ W, int ldw,
    const float* __restrict__ bias,
    float* __restrict__ Cf, unsigned short* __restrict__ Cb, int ldc,
    int M, int N, int K, int relu)
{
    __shared__ unsigned short sA[128][32];
    __shared__ unsigned int   sB[16][64];

    const int tid  = threadIdx.x;
    const int lane = tid & 31;
    const int wv   = tid >> 5;
    const int wm   = wv >> 1, wn = wv & 1;     // 4x2 wave grid, 32x32 each
    const int m0   = blockIdx.y * 128;
    const int n0   = blockIdx.x * 64;
    (void)M; (void)N;

    v8f acc[2][2];
#pragma unroll
    for (int i = 0; i < 2; ++i)
#pragma unroll
        for (int j = 0; j < 2; ++j) acc[i][j] = (v8f){0,0,0,0,0,0,0,0};

    for (int k0 = 0; k0 < K; k0 += 32) {
#pragma unroll
        for (int i = 0; i < 2; ++i) {          // A tile: 128x32, 16B per copy
            int idx = tid + i * 256;
            int row = idx >> 2, seg = idx & 3;
            *(uint4*)&sA[row][seg * 8] =
                *(const uint4*)(A + (size_t)(m0 + row) * lda + k0 + seg * 8);
        }
#pragma unroll
        for (int i = 0; i < 4; ++i) {          // B tile, packed by K-pairs
            int idx = tid + i * 256;
            int r2 = idx >> 6, c = idx & 63;
            int kk = k0 + (r2 << 1);
            unsigned int lo = W[(size_t)kk * ldw + n0 + c];
            unsigned int hi = W[(size_t)(kk + 1) * ldw + n0 + c];
            sB[r2][c] = lo | (hi << 16);
        }
        if (k0 + 32 < K) {                     // cover latency of next tile
            __builtin_prefetch(A + (size_t)(m0 + (tid >> 1)) * lda + k0 + 32, 0, 0);
            __builtin_prefetch(W + (size_t)(k0 + 32 + (tid >> 4)) * ldw +
                               n0 + (tid & 15) * 4, 0, 0);
        }
        __syncthreads();

        v8u a0 = load_frag_A(&sA[0][0], 32, wm * 32, 0);
        v8u a1 = load_frag_A(&sA[0][0], 32, wm * 32 + 16, 0);
        v8u b0 = load_frag_Bp(&sB[0][0], 64, 0, wn * 32);
        v8u b1 = load_frag_Bp(&sB[0][0], 64, 0, wn * 32 + 16);
        acc[0][0] = wmma_bf16(a0, b0, acc[0][0]);
        acc[0][1] = wmma_bf16(a0, b1, acc[0][1]);
        acc[1][0] = wmma_bf16(a1, b0, acc[1][0]);
        acc[1][1] = wmma_bf16(a1, b1, acc[1][1]);
        __syncthreads();
    }

#pragma unroll
    for (int tm = 0; tm < 2; ++tm)
#pragma unroll
    for (int tn = 0; tn < 2; ++tn) {
        v8f a = acc[tm][tn];
        int col = n0 + wn * 32 + tn * 16 + (lane & 15);
        float bv = bias ? bias[col] : 0.f;
#pragma unroll
        for (int r = 0; r < 8; ++r) {
            int row = m0 + wm * 32 + tm * 16 + r + ((lane >> 4) << 3);
            float v = a[r] + bv;
            if (relu) v = fmaxf(v, 0.f);
            if (Cf) Cf[(size_t)row * ldc + col] = v;
            if (Cb) Cb[(size_t)row * ldc + col] = f32_to_bf16(v);
        }
    }
}

// ---------------------------------------------------------------------------
// Fused attention per (b,h). MODE: 0 = char-pos + [q,k] mask (self, causal),
// 1 = char-pos + [b,k] mask, 2 = word-pos (qpos[b]) + [b,k] mask.
// V tile is DMA'd to LDS by the Tensor Data Mover (issued by wave 0) and
// overlapped with the Q@emb^T and Q@K^T WMMA phases; drained with
// s_wait_tensorcnt before the P@V phase.
// ---------------------------------------------------------------------------
template <int SK, int MODE>
__global__ __launch_bounds__(128) void attn_kernel(
    const unsigned short* __restrict__ Qb, int qrs, int qhm,
    const unsigned short* __restrict__ Kb, int krs, int khm,
    const unsigned short* __restrict__ Vb, int vrs, int vhm,
    const unsigned short* __restrict__ embPad,      // [144][64] bf16
    const unsigned char* __restrict__ maskQK,       // [64][SK] or null
    const unsigned char* __restrict__ maskBK,       // [B][SK] or null
    const int* __restrict__ qpos,                   // [B] or null
    unsigned short* __restrict__ ctx, int ostride,
    float scale)
{
    constexpr int NKT = SK / 16;
    constexpr int NKS = SK / 32;
    __shared__ unsigned short sP[64][SK];           // softmax probs (bf16)
    __shared__ unsigned short sVr[SK][64];          // V tile (row-major, TDM)
    __shared__ unsigned short sPq[4][16][144];      // pos logits per strip

    const int h = blockIdx.x;
    const int b = blockIdx.y;
    const int tid  = threadIdx.x;
    const int lane = tid & 31;
    const int w    = tid >> 5;        // wave -> 16-row strip
    const int q0   = w * 16;
    const int n    = lane & 15;
    const int hb   = lane >> 4;

    // 1) kick off async TDM copy of V[SK][64] into LDS (wave 0 only; the
    //    DMA overlaps with the WMMA phases below).
    if (w == 0) {
        const unsigned short* V0 = Vb + (size_t)b * SK * vrs + h * vhm;
        tdm_load_tile_2d(V0, SK, 64, (unsigned)vrs, &sVr[0][0]);
    }

    // 2) Q fragments for this strip (d = 0..31, 32..63)
    const unsigned short* Q0 = Qb + (size_t)b * 64 * qrs + h * qhm;
    v8u qa0 = load_frag_A(Q0, qrs, q0, 0);
    v8u qa1 = load_frag_A(Q0, qrs, q0, 32);

    // 3) Pq = Q @ emb^T  (strip-private, stored bf16 in LDS)
#pragma unroll
    for (int jt = 0; jt < 9; ++jt) {
        v8f acc = {0,0,0,0,0,0,0,0};
        v8u e0 = load_frag_BT(embPad, 64, jt * 16, 0);
        acc = wmma_bf16(qa0, e0, acc);
        v8u e1 = load_frag_BT(embPad, 64, jt * 16, 32);
        acc = wmma_bf16(qa1, e1, acc);
#pragma unroll
        for (int r = 0; r < 8; ++r)
            sPq[w][r + (hb << 3)][jt * 16 + n] = f32_to_bf16(acc[r]);
    }

    // 4) S = Q @ K^T
    v8f sacc[NKT];
    const unsigned short* K0 = Kb + (size_t)b * SK * krs + h * khm;
#pragma unroll
    for (int kt = 0; kt < NKT; ++kt) {
        v8f acc = {0,0,0,0,0,0,0,0};
        v8u kb0 = load_frag_BT(K0, krs, kt * 16, 0);
        acc = wmma_bf16(qa0, kb0, acc);
        v8u kb1 = load_frag_BT(K0, krs, kt * 16, 32);
        acc = wmma_bf16(qa1, kb1, acc);
        sacc[kt] = acc;
    }

    const int qp = (MODE == 2) ? qpos[b] : 0;

    // 5) logits + row softmax (row = VGPR index; cols live across 16 lanes)
#pragma unroll
    for (int r = 0; r < 8; ++r) {
        int m = r + (hb << 3);
        int q = q0 + m;
        float mx = -3.0e38f;
#pragma unroll
        for (int kt = 0; kt < NKT; ++kt) {
            int k = kt * 16 + n;
            int rel = (MODE == 2) ? (k - qp) : (k - q);
            rel = rel < -64 ? -64 : (rel > 64 ? 64 : rel);
            float pos = bf16_to_f32(sPq[w][m][rel + 64]);
            float v = (sacc[kt][r] + pos) * scale;
            bool msk;
            if constexpr (MODE == 0) msk = maskQK[q * SK + k] != 0;
            else                     msk = maskBK[(size_t)b * SK + k] != 0;
            if (msk) v = -1.0e30f;
            sacc[kt][r] = v;
            mx = fmaxf(mx, v);
        }
#pragma unroll
        for (int d = 1; d < 16; d <<= 1) mx = fmaxf(mx, __shfl_xor(mx, d, 32));
        float sum = 0.f;
#pragma unroll
        for (int kt = 0; kt < NKT; ++kt) {
            float e = __expf(sacc[kt][r] - mx);
            sacc[kt][r] = e;
            sum += e;
        }
#pragma unroll
        for (int d = 1; d < 16; d <<= 1) sum += __shfl_xor(sum, d, 32);
        float inv = 1.f / sum;
#pragma unroll
        for (int kt = 0; kt < NKT; ++kt)
            sP[q0 + m][kt * 16 + n] = f32_to_bf16(sacc[kt][r] * inv);
    }

    // 6) drain the V DMA (wave 0's TENSORcnt), then publish via barrier
    if (w == 0) __builtin_amdgcn_s_wait_tensorcnt(0);
    __syncthreads();

    // 7) O = P @ V  -> ctx (bf16, layout [b*64+q][h*64+d])
#pragma unroll
    for (int nt = 0; nt < 4; ++nt) {
        v8f acc = {0,0,0,0,0,0,0,0};
#pragma unroll
        for (int ks = 0; ks < NKS; ++ks) {
            v8u pa = load_frag_A(&sP[0][0], SK, q0, ks * 32);
            v8u vb = load_frag_Brow(&sVr[0][0], 64, ks * 32, nt * 16);
            acc = wmma_bf16(pa, vb, acc);
        }
        int col = h * 64 + nt * 16 + n;
#pragma unroll
        for (int r = 0; r < 8; ++r) {
            int q = q0 + r + (hb << 3);
            ctx[((size_t)b * 64 + q) * ostride + col] = f32_to_bf16(acc[r]);
        }
    }
}

// ---------------------------------------------------------------------------
// LayerNorm of (a + res), row length 512; writes f32 and/or bf16.
// ---------------------------------------------------------------------------
__global__ __launch_bounds__(128) void ln_kernel(
    const float* __restrict__ a, const float* __restrict__ res,
    const float* __restrict__ g, const float* __restrict__ beta,
    float* __restrict__ outF, unsigned short* __restrict__ outB)
{
    __shared__ float red[128];
    const int row = blockIdx.x;
    const int tid = threadIdx.x;
    float vals[4];
    float s = 0.f;
#pragma unroll
    for (int i = 0; i < 4; ++i) {
        int c = tid + i * 128;
        float v = a[(size_t)row * 512 + c];
        if (res) v += res[(size_t)row * 512 + c];
        vals[i] = v; s += v;
    }
    red[tid] = s; __syncthreads();
    for (int st = 64; st > 0; st >>= 1) {
        if (tid < st) red[tid] += red[tid + st];
        __syncthreads();
    }
    float mean = red[0] * (1.f / 512.f);
    __syncthreads();
    float vs = 0.f;
#pragma unroll
    for (int i = 0; i < 4; ++i) { float d = vals[i] - mean; vs += d * d; }
    red[tid] = vs; __syncthreads();
    for (int st = 64; st > 0; st >>= 1) {
        if (tid < st) red[tid] += red[tid + st];
        __syncthreads();
    }
    float rstd = rsqrtf(red[0] * (1.f / 512.f) + 1e-5f);
#pragma unroll
    for (int i = 0; i < 4; ++i) {
        int c = tid + i * 128;
        float y = (vals[i] - mean) * rstd * g[c] + beta[c];
        if (outF) outF[(size_t)row * 512 + c] = y;
        if (outB) outB[(size_t)row * 512 + c] = f32_to_bf16(y);
    }
}

// ---------------------------------------------------------------------------
// Host orchestration
// ---------------------------------------------------------------------------
extern "C" void kernel_launch(void* const* d_in, const int* in_sizes, int n_in,
                              void* d_out, int out_size, void* d_ws, size_t ws_size,
                              hipStream_t stream)
{
    (void)in_sizes; (void)n_in; (void)out_size; (void)ws_size;

    const int M1 = 512 * 64;    // tokens  (B*SQ) = 32768
    const int MC = 512 * 128;   // char rows      = 65536
    const int MW = 512 * 32;    // word rows      = 16384

    const float* self_input = (const float*)d_in[0];
    const float* a1_bqkv = (const float*)d_in[4];
    const float* a1_bo   = (const float*)d_in[6];
    const float* a2_bq   = (const float*)d_in[9];
    const float* a2_bk   = (const float*)d_in[11];
    const float* a2_bv   = (const float*)d_in[13];
    const float* a2_bo   = (const float*)d_in[15];
    const float* a3_bq   = (const float*)d_in[18];
    const float* a3_bk   = (const float*)d_in[20];
    const float* a3_bv   = (const float*)d_in[22];
    const float* a3_bo   = (const float*)d_in[24];
    const float* ffn_b1  = (const float*)d_in[27];
    const float* ffn_b2  = (const float*)d_in[29];
    const float* ln1_g = (const float*)d_in[30];
    const float* ln1_b = (const float*)d_in[31];
    const float* ln2_g = (const float*)d_in[32];
    const float* ln2_b = (const float*)d_in[33];
    const float* ln3_g = (const float*)d_in[34];
    const float* ln3_b = (const float*)d_in[35];
    const unsigned char* look_mask = (const unsigned char*)d_in[36];
    const unsigned char* char_mask = (const unsigned char*)d_in[37];
    const unsigned char* word_mask = (const unsigned char*)d_in[38];
    const int* sent_lens = (const int*)d_in[39];

    char* ws = (char*)d_ws;
    size_t off = 0;
    auto alloc = [&](size_t bytes) {
        size_t o = off; off += (bytes + 255) & ~(size_t)255; return o;
    };
    unsigned short* XBF = (unsigned short*)(ws + alloc((size_t)M1 * 512 * 2));
    float*          X   = (float*)        (ws + alloc((size_t)M1 * 512 * 4));
    unsigned short* CHB = (unsigned short*)(ws + alloc((size_t)MC * 512 * 2));
    unsigned short* WDB = (unsigned short*)(ws + alloc((size_t)MW * 512 * 2));
    unsigned short* BIG = (unsigned short*)(ws + alloc((size_t)M1 * 2048 * 2));
    unsigned short* QB  = (unsigned short*)(ws + alloc((size_t)M1 * 512 * 2));
    unsigned short* KB  = (unsigned short*)(ws + alloc((size_t)MC * 512 * 2));
    unsigned short* VB  = (unsigned short*)(ws + alloc((size_t)MC * 512 * 2));
    unsigned short* CTX = (unsigned short*)(ws + alloc((size_t)M1 * 512 * 2));
    float*          PRJ = (float*)        (ws + alloc((size_t)M1 * 512 * 4));
    unsigned short* WB  = (unsigned short*)(ws + alloc((size_t)5242880 * 2));
    unsigned short* EMB1= (unsigned short*)(ws + alloc((size_t)144 * 64 * 2));
    unsigned short* EMB2= (unsigned short*)(ws + alloc((size_t)144 * 64 * 2));
    unsigned short* EMB3= (unsigned short*)(ws + alloc((size_t)144 * 64 * 2));
    int*            QPS = (int*)          (ws + alloc((size_t)512 * 4));

    auto cvt = [&](const float* src, unsigned short* dst, size_t nE) {
        cvt_bf16_kernel<<<dim3((unsigned)((nE + 255) / 256)), 256, 0, stream>>>(
            src, dst, (int)nE);
    };
    auto gemm = [&](const unsigned short* A, int lda, const unsigned short* W,
                    int ldw, const float* bias, float* Cf, unsigned short* Cb,
                    int ldc, int M, int N, int K, int relu) {
        gemm_bf16_kernel<<<dim3(N / 64, M / 128), 256, 0, stream>>>(
            A, lda, W, ldw, bias, Cf, Cb, ldc, M, N, K, relu);
    };

    // --- weight conversion ---
    const int    widx[12] = {3, 5, 8, 10, 12, 14, 17, 19, 21, 23, 26, 28};
    const size_t wcnt[12] = {512u*1536, 512u*512, 512u*512, 512u*512, 512u*512,
                             512u*512, 512u*512, 512u*512, 512u*512, 512u*512,
                             512u*2048, 2048u*512};
    unsigned short* wp[12];
    {
        size_t a = 0;
        for (int i = 0; i < 12; ++i) {
            wp[i] = WB + a; a += wcnt[i];
            cvt((const float*)d_in[widx[i]], wp[i], wcnt[i]);
        }
    }
    unsigned short *w_qkv = wp[0], *w_a1o = wp[1];
    unsigned short *w_a2q = wp[2], *w_a2k = wp[3], *w_a2v = wp[4], *w_a2o = wp[5];
    unsigned short *w_a3q = wp[6], *w_a3k = wp[7], *w_a3v = wp[8], *w_a3o = wp[9];
    unsigned short *w_f1 = wp[10], *w_f2 = wp[11];

    // --- activation conversion + prep ---
    cvt(self_input, XBF, (size_t)M1 * 512);
    cvt((const float*)d_in[1], CHB, (size_t)MC * 512);
    cvt((const float*)d_in[2], WDB, (size_t)MW * 512);
    prep_emb_kernel<<<36, 256, 0, stream>>>((const float*)d_in[7],  EMB1);
    prep_emb_kernel<<<36, 256, 0, stream>>>((const float*)d_in[16], EMB2);
    prep_emb_kernel<<<36, 256, 0, stream>>>((const float*)d_in[25], EMB3);
    prep_qpos_kernel<<<1, 128, 0, stream>>>(sent_lens, 16, 512, QPS);

    const float SCALE_INV = 0.125f;   // 1/sqrt(64)
    dim3 agrid(8, 512);               // (H, B)

    // --- sublayer 1: self attention (fused QKV) ---
    gemm(XBF, 512, w_qkv, 1536, a1_bqkv, nullptr, BIG, 1536, M1, 1536, 512, 0);
    attn_kernel<64, 0><<<agrid, 128, 0, stream>>>(
        BIG, 1536, 192, BIG + 64, 1536, 192, BIG + 128, 1536, 192,
        EMB1, look_mask, nullptr, nullptr, CTX, 512, SCALE_INV);
    gemm(CTX, 512, w_a1o, 512, a1_bo, PRJ, nullptr, 512, M1, 512, 512, 0);
    ln_kernel<<<M1, 128, 0, stream>>>(PRJ, self_input, ln1_g, ln1_b, X, XBF);

    // --- sublayer 2: cross attention to char_enc (Sk=128) ---
    gemm(XBF, 512, w_a2q, 512, a2_bq, nullptr, QB, 512, M1, 512, 512, 0);
    gemm(CHB, 512, w_a2k, 512, a2_bk, nullptr, KB, 512, MC, 512, 512, 0);
    gemm(CHB, 512, w_a2v, 512, a2_bv, nullptr, VB, 512, MC, 512, 512, 0);
    attn_kernel<128, 1><<<agrid, 128, 0, stream>>>(
        QB, 512, 64, KB, 512, 64, VB, 512, 64,
        EMB2, nullptr, char_mask, nullptr, CTX, 512, SCALE_INV);
    gemm(CTX, 512, w_a2o, 512, a2_bo, PRJ, nullptr, 512, M1, 512, 512, 0);
    ln_kernel<<<M1, 128, 0, stream>>>(PRJ, X, ln2_g, ln2_b, X, XBF);

    // --- sublayer 3: cross attention to word_enc (Sk=32, word positions) ---
    gemm(XBF, 512, w_a3q, 512, a3_bq, nullptr, QB, 512, M1, 512, 512, 0);
    gemm(WDB, 512, w_a3k, 512, a3_bk, nullptr, KB, 512, MW, 512, 512, 0);
    gemm(WDB, 512, w_a3v, 512, a3_bv, nullptr, VB, 512, MW, 512, 512, 0);
    attn_kernel<32, 2><<<agrid, 128, 0, stream>>>(
        QB, 512, 64, KB, 512, 64, VB, 512, 64,
        EMB3, nullptr, word_mask, QPS, CTX, 512, SCALE_INV);
    gemm(CTX, 512, w_a3o, 512, a3_bo, PRJ, nullptr, 512, M1, 512, 512, 0);
    ln_kernel<<<M1, 128, 0, stream>>>(PRJ, X, ln3_g, ln3_b, X, XBF);

    // --- FFN + final LN -> d_out (f32) ---
    gemm(XBF, 512, w_f1, 2048, ffn_b1, nullptr, BIG, 2048, M1, 2048, 512, 1);
    gemm(BIG, 2048, w_f2, 512, ffn_b2, PRJ, nullptr, 512, M1, 512, 2048, 0);
    ln_kernel<<<M1, 128, 0, stream>>>(PRJ, X, ln3_g, ln3_b, (float*)d_out, nullptr);
}
